// MoE_68771016343958
// MI455X (gfx1250) — matrix operand order repaired and checked by reading, not compiled, not run on hardware
//
#include <hip/hip_runtime.h>
#include <hip/hip_bf16.h>
#include <math.h>

// ---------------------------------------------------------------------------
// MoE (top-2, 8 experts + dummy identity) for gfx1250 (MI455X, wave32, WMMA).
// Routed (gathered) computation: only top-2 (token,expert) pairs are computed,
// ~137 GFLOP instead of the reference's dense 550 GFLOP.
// bf16 V_WMMA_F32_16X16X32_BF16 is the compute path; weights are converted
// once per call into pre-swizzled bf16 WMMA B-fragments (512 elems = 1 KB per
// fragment; packed weights = 2 x 67 MB in ws).
// ---------------------------------------------------------------------------

#define Bk 2
#define Tk 2048
#define Nk (Bk * Tk)        // 4096 tokens
#define Dk 1024
#define Fk 4096
#define Ek 8                // real experts (index Ek == dummy identity)
#define ROWS 64             // tokens per FFN tile
#define FC 128              // F-chunk width (multiple of 32)
#define MAX_TILES (2 * Nk / ROWS + Ek)   // 136
#define PADN (2 * Nk + Ek * ROWS)        // 8704 gather slots (64-padded)
#define FRAG 512            // elements per packed B fragment (32 lanes x 16)

typedef __attribute__((ext_vector_type(16))) __bf16 v16bf;
typedef __attribute__((ext_vector_type(8)))  __bf16 v8bf;
typedef __attribute__((ext_vector_type(8)))  float  v8f;

// ---------------- fragment helpers (contiguous k = 16*(lane>>4) + j, same order in A and B) ----------

// A-matrix 16x32 bf16 fragment from an LDS tile (row-major, `stride` elems):
// lane<16 holds row=lane, K in {kb+0..15}; lane>=16 row=lane-16,
// K in {kb+16..31}. Two contiguous 16B chunks per lane.
__device__ inline v16bf load_a_frag_lds(const __bf16* base, int stride,
                                        int rowbase, int kbase, int lane) {
  int row = rowbase + (lane & 15);
  int kh  = (lane >> 4) << 4;
  const __bf16* p0 = base + row * stride + kbase + kh;
  v8bf lo = *reinterpret_cast<const v8bf*>(p0);
  v8bf hi = *reinterpret_cast<const v8bf*>(p0 + 8);
  return __builtin_shufflevector(lo, hi, 0, 1, 2, 3, 4, 5, 6, 7,
                                 8, 9, 10, 11, 12, 13, 14, 15);
}

// B fragments are pre-packed in ws: each fragment = 32 lanes * 16 bf16 (1 KB),
// lane holds col = lane&15, K = (lane>>4)*16 + j  (j contiguous).
__device__ inline v16bf load_b_frag(const __bf16* fragbase, int lane) {
  union { uint4 u[2]; v16bf v; } t;
  const uint4* p = reinterpret_cast<const uint4*>(fragbase + lane * 16);
  t.u[0] = p[0];
  t.u[1] = p[1];
  return t.v;
}

// Branch-free erf (Abramowitz-Stegun 7.1.26, max err ~1.5e-7): keeps the
// epilogue between WMMA phases straight-line VALU (no EXEC divergence, unlike
// libm erff which splits on |x|<1).
__device__ inline float erf_fast(float v) {
  float a = fabsf(v);
  float t = __frcp_rn(fmaf(0.3275911f, a, 1.0f));
  float p = fmaf(1.061405429f, t, -1.453152027f);
  p = fmaf(p, t, 1.421413741f);
  p = fmaf(p, t, -0.284496736f);
  p = fmaf(p, t, 0.254829592f);
  float y = 1.0f - p * t * __expf(-a * a);
  return copysignf(y, v);
}

__device__ inline float gelu_exact(float v) {
  return 0.5f * v * (1.0f + erf_fast(v * 0.70710678118654752f));
}

// ---------------- kernel 1: router (logits -> softmax -> top-2) ------------

__global__ __launch_bounds__(256) void router_kernel(
    const float* __restrict__ x, const float* __restrict__ rw,
    const float* __restrict__ rb, int* __restrict__ ti,
    float* __restrict__ tp, float* __restrict__ g9) {
  int wave = threadIdx.x >> 5, lane = threadIdx.x & 31;
  int t = blockIdx.x * 8 + wave;  // one wave32 per token
  float acc[Ek + 1];
#pragma unroll
  for (int e = 0; e <= Ek; ++e) acc[e] = 0.f;
  const float* xr = x + (size_t)t * Dk;
  for (int d = lane; d < Dk; d += 32) {
    float xv = xr[d];
    const float* w = rw + d * (Ek + 1);
#pragma unroll
    for (int e = 0; e <= Ek; ++e) acc[e] += xv * w[e];
  }
#pragma unroll
  for (int e = 0; e <= Ek; ++e) {
    float v = acc[e];
    for (int off = 16; off > 0; off >>= 1) v += __shfl_xor(v, off, 32);
    acc[e] = v;
  }
  if (lane == 0) {
    float lg[Ek + 1];
    float m = -1e30f;
    for (int e = 0; e <= Ek; ++e) { lg[e] = acc[e] + rb[e]; m = fmaxf(m, lg[e]); }
    float s = 0.f;
    for (int e = 0; e <= Ek; ++e) { lg[e] = __expf(lg[e] - m); s += lg[e]; }
    float inv = 1.f / s;
    int i1 = -1, i2 = -1;
    float p1 = -1.f, p2 = -1.f;
    for (int e = 0; e <= Ek; ++e) {   // '>' keeps lowest index on ties (jax)
      float p = lg[e] * inv;
      if (p > p1)      { i2 = i1; p2 = p1; i1 = e; p1 = p; }
      else if (p > p2) { i2 = e; p2 = p; }
    }
    ti[t * 2 + 0] = i1; ti[t * 2 + 1] = i2;
    tp[t * 2 + 0] = p1; tp[t * 2 + 1] = p2;
    g9[t] = (i1 == Ek) ? p1 : ((i2 == Ek) ? p2 : 0.f);
  }
}

// ---------------- kernel 2a/2b: weight f32 -> bf16 WMMA-fragment packing ---

// w1: [E, D, F] (K-major = D).  One wave per 16x32 fragment.
__global__ __launch_bounds__(256) void pack_w1_kernel(
    const float* __restrict__ w1, __bf16* __restrict__ w1p) {
  int gw = blockIdx.x * 8 + (threadIdx.x >> 5);  // frag id, e*8192 + ct*32 + kt
  int lane = threadIdx.x & 31;
  int e = gw >> 13;
  int rem = gw & 8191;
  int ct = rem >> 5;            // F col-tile (0..255)
  int kt = rem & 31;            // D k-tile (0..31)
  int col = ct * 16 + (lane & 15);
  int kbase = kt * 32 + ((lane >> 4) << 4);
  const float* src = w1 + (size_t)e * Dk * Fk;
  union { __bf16 b[16]; uint4 u[2]; } tmp;
#pragma unroll
  for (int j = 0; j < 16; ++j)
    tmp.b[j] = (__bf16)src[(size_t)(kbase + j) * Fk + col];
  uint4* dst = reinterpret_cast<uint4*>(w1p + (size_t)gw * FRAG + lane * 16);
  dst[0] = tmp.u[0];
  dst[1] = tmp.u[1];
}

// w2: [E, F, D] (K-major = F).
__global__ __launch_bounds__(256) void pack_w2_kernel(
    const float* __restrict__ w2, __bf16* __restrict__ w2p) {
  int gw = blockIdx.x * 8 + (threadIdx.x >> 5);  // e*8192 + dct*128 + kt
  int lane = threadIdx.x & 31;
  int e = gw >> 13;
  int rem = gw & 8191;
  int dct = rem >> 7;           // D col-tile (0..63)
  int kt = rem & 127;           // F k-tile (0..127)
  int col = dct * 16 + (lane & 15);
  int kbase = kt * 32 + ((lane >> 4) << 4);
  const float* src = w2 + (size_t)e * Fk * Dk;
  union { __bf16 b[16]; uint4 u[2]; } tmp;
#pragma unroll
  for (int j = 0; j < 16; ++j)
    tmp.b[j] = (__bf16)src[(size_t)(kbase + j) * Dk + col];
  uint4* dst = reinterpret_cast<uint4*>(w2p + (size_t)gw * FRAG + lane * 16);
  dst[0] = tmp.u[0];
  dst[1] = tmp.u[1];
}

// ---------------- kernel 3: deterministic gather (counts+scan+scatter) ----

__global__ __launch_bounds__(256) void scan_kernel(
    const int* __restrict__ ti, const float* __restrict__ tp,
    int* __restrict__ gtok, float* __restrict__ gprob,
    int* __restrict__ tile_expert, int* __restrict__ tile_base,
    int* __restrict__ total_tiles) {
  __shared__ int cnt[Ek];
  __shared__ int off[Ek];
  int wave = threadIdx.x >> 5, lane = threadIdx.x & 31;
  if (wave < Ek) {
    int c = 0;
    for (int base = 0; base < Nk; base += 32) {
      int t = base + lane;
      c += (ti[t * 2] == wave) + (ti[t * 2 + 1] == wave);
    }
    for (int o = 16; o > 0; o >>= 1) c += __shfl_xor(c, o, 32);
    if (lane == 0) cnt[wave] = c;
  }
  __syncthreads();
  if (threadIdx.x == 0) {
    int o = 0, tl = 0;
    for (int e = 0; e < Ek; ++e) {
      off[e] = o;
      int nt = (cnt[e] + ROWS - 1) / ROWS;
      for (int i = 0; i < nt; ++i) {
        tile_expert[tl] = e;
        tile_base[tl] = o + i * ROWS;
        ++tl;
      }
      o += nt * ROWS;
    }
    *total_tiles = tl;
  }
  __syncthreads();
  if (wave < Ek) {
    int o = off[wave];
    for (int base = 0; base < Nk; base += 32) {
      int t = base + lane;
      int a = ti[t * 2], b = ti[t * 2 + 1];
      bool m = (a == wave) || (b == wave);
      float p = (a == wave) ? tp[t * 2] : tp[t * 2 + 1];
      unsigned mask = (unsigned)__ballot(m);
      if (m) {
        int rank = __popc(mask & ((1u << lane) - 1u));
        gtok[o + rank] = t;
        gprob[o + rank] = p;
      }
      o += __popc(mask);
    }
    int padded_end = off[wave] + ((cnt[wave] + ROWS - 1) / ROWS) * ROWS;
    for (int i = off[wave] + cnt[wave] + lane; i < padded_end; i += 32) {
      gtok[i] = 0;       // safe dummy row
      gprob[i] = 0.f;    // zero contribution
    }
  }
}

// ---------------- kernel 4: out = dummy_gate * x ---------------------------

__global__ __launch_bounds__(256) void out_init_kernel(
    const float* __restrict__ x, const float* __restrict__ g9,
    float* __restrict__ out) {
  size_t i = (size_t)blockIdx.x * blockDim.x + threadIdx.x;  // over Nk*Dk/4
  int r = (int)(i >> 8);  // Dk/4 == 256 float4 per row
  float g = g9[r];
  float4 v = reinterpret_cast<const float4*>(x)[i];
  float4 o;
  o.x = g * v.x; o.y = g * v.y; o.z = g * v.z; o.w = g * v.w;
  reinterpret_cast<float4*>(out)[i] = o;
}

// ---------------- kernel 5: fused expert FFN (WMMA bf16) -------------------
// One 64-token tile per workgroup, 512 threads = 16 waves.
// LDS: X tile 128 KB + H chunk 16 KB (CDNA5 allows up to 320 KB / WG).

__global__ __launch_bounds__(512) void ffn_kernel(
    const float* __restrict__ x, const __bf16* __restrict__ w1p,
    const __bf16* __restrict__ w2p, const int* __restrict__ tile_expert,
    const int* __restrict__ tile_base, const int* __restrict__ total_tiles,
    const int* __restrict__ gtok, const float* __restrict__ gprob,
    float* __restrict__ out) {
  __shared__ __bf16 Xs[ROWS * Dk];
  __shared__ __bf16 Hs[ROWS * FC];
  __shared__ int toks[ROWS];
  __shared__ float prbs[ROWS];

  if (blockIdx.x >= *total_tiles) return;  // uniform block-wide exit
  int e = tile_expert[blockIdx.x];
  int base = tile_base[blockIdx.x];
  int tid = threadIdx.x, wave = tid >> 5, lane = tid & 31;

  if (tid < ROWS) { toks[tid] = gtok[base + tid]; prbs[tid] = gprob[base + tid]; }
  __syncthreads();

  // Stage gathered X rows into LDS as bf16.
  for (int i = tid; i < ROWS * (Dk / 4); i += 512) {
    int r = i >> 8;         // Dk/4 = 256
    int c4 = i & 255;
    float4 v = reinterpret_cast<const float4*>(x)[(size_t)toks[r] * (Dk / 4) + c4];
    union { __bf16 b[4]; uint2 u; } pk;
    pk.b[0] = (__bf16)v.x; pk.b[1] = (__bf16)v.y;
    pk.b[2] = (__bf16)v.z; pk.b[3] = (__bf16)v.w;
    *reinterpret_cast<uint2*>(&Xs[(r << 10) + (c4 << 2)]) = pk.u;
  }
  __syncthreads();

  int rt = wave & 3;   // row tile (of 4)
  int wq = wave >> 2;  // 0..3: column strip

  v8f acc[16] = {};    // y[64,1024]: 16 16x16 tiles per wave, persistent

  for (int fc = 0; fc < Fk / FC; ++fc) {
    // ---- phase 1: H = gelu(X @ W1[:, chunk]) ----
    v8f hacc[2] = {};
    for (int kt = 0; kt < Dk / 32; ++kt) {
      v16bf a = load_a_frag_lds(Xs, Dk, rt * 16, kt * 32, lane);
#pragma unroll
      for (int i = 0; i < 2; ++i) {
        int ctg = fc * (FC / 16) + wq * 2 + i;
        const __bf16* fb =
            w1p + (((size_t)e * (Fk / 16) + ctg) * (Dk / 32) + kt) * FRAG;
        __builtin_prefetch(fb + FRAG, 0, 1);  // next k-tile fragment
        v16bf b = load_b_frag(fb, lane);
        hacc[i] = __builtin_amdgcn_wmma_f32_16x16x32_bf16(
            false, a, false, b, (short)0, hacc[i], false, false);
      }
    }
#pragma unroll
    for (int i = 0; i < 2; ++i) {
      int ctl = wq * 2 + i;
#pragma unroll
      for (int r = 0; r < 8; ++r) {
        int M = rt * 16 + ((lane >> 4) << 3) + r;
        int Nc = ctl * 16 + (lane & 15);
        Hs[M * FC + Nc] = (__bf16)gelu_exact(hacc[i][r]);
      }
    }
    __syncthreads();
    // ---- phase 2: Y += H @ W2[chunk, :] ----
    for (int kt2 = 0; kt2 < FC / 32; ++kt2) {
      v16bf a2 = load_a_frag_lds(Hs, FC, rt * 16, kt2 * 32, lane);
      int ktg = fc * (FC / 32) + kt2;
#pragma unroll
      for (int i = 0; i < 16; ++i) {
        int dct = wq * 16 + i;
        const __bf16* fb =
            w2p + (((size_t)e * (Dk / 16) + dct) * (Fk / 32) + ktg) * FRAG;
        v16bf b = load_b_frag(fb, lane);
        acc[i] = __builtin_amdgcn_wmma_f32_16x16x32_bf16(
            false, a2, false, b, (short)0, acc[i], false, false);
      }
    }
    __syncthreads();
  }

  // ---- epilogue: out[tok] += gate * y ----
#pragma unroll
  for (int i = 0; i < 16; ++i) {
    int dct = wq * 16 + i;
#pragma unroll
    for (int r = 0; r < 8; ++r) {
      int M = rt * 16 + ((lane >> 4) << 3) + r;
      int col = dct * 16 + (lane & 15);
      atomicAdd(&out[(size_t)toks[M] * Dk + col], prbs[M] * acc[i][r]);
    }
  }
}

// ---------------- host launcher --------------------------------------------

extern "C" void kernel_launch(void* const* d_in, const int* in_sizes, int n_in,
                              void* d_out, int out_size, void* d_ws,
                              size_t ws_size, hipStream_t stream) {
  (void)in_sizes; (void)n_in; (void)out_size;
  const float* x  = (const float*)d_in[0];
  const float* rw = (const float*)d_in[1];
  const float* rb = (const float*)d_in[2];
  const float* w1 = (const float*)d_in[3];
  const float* w2 = (const float*)d_in[4];
  float* out = (float*)d_out;

  // Workspace carve-up (~135 MB; ws_size assumed sufficient).
  char* ws = (char*)d_ws;
  size_t off = 0;
  auto alloc = [&](size_t bytes) {
    off = (off + 255) & ~(size_t)255;
    size_t r = off;
    off += bytes;
    return r;
  };
  int*    ti   = (int*)  (ws + alloc((size_t)Nk * 2 * sizeof(int)));
  float*  tp   = (float*)(ws + alloc((size_t)Nk * 2 * sizeof(float)));
  float*  g9   = (float*)(ws + alloc((size_t)Nk * sizeof(float)));
  int*    gtok = (int*)  (ws + alloc((size_t)PADN * sizeof(int)));
  float*  gprb = (float*)(ws + alloc((size_t)PADN * sizeof(float)));
  int*    texp = (int*)  (ws + alloc((size_t)MAX_TILES * sizeof(int)));
  int*    tbas = (int*)  (ws + alloc((size_t)MAX_TILES * sizeof(int)));
  int*    ttot = (int*)  (ws + alloc(sizeof(int)));
  __bf16* w1p  = (__bf16*)(ws + alloc((size_t)Ek * Dk * Fk * sizeof(__bf16)));
  __bf16* w2p  = (__bf16*)(ws + alloc((size_t)Ek * Fk * Dk * sizeof(__bf16)));
  (void)ws_size;

  router_kernel<<<Nk / 8, 256, 0, stream>>>(x, rw, rb, ti, tp, g9);
  pack_w1_kernel<<<(Ek * (Fk / 16) * (Dk / 32)) / 8, 256, 0, stream>>>(w1, w1p);
  pack_w2_kernel<<<(Ek * (Dk / 16) * (Fk / 32)) / 8, 256, 0, stream>>>(w2, w2p);
  scan_kernel<<<1, 256, 0, stream>>>(ti, tp, gtok, gprb, texp, tbas, ttot);
  out_init_kernel<<<(Nk * (Dk / 4)) / 256, 256, 0, stream>>>(x, g9, out);
  ffn_kernel<<<MAX_TILES, 512, 0, stream>>>(x, w1p, w2p, texp, tbas, ttot,
                                            gtok, gprb, out);
}
